// MTREncoder_58703613002025
// MI455X (gfx1250) — matrix operand order repaired and checked by reading, not compiled, not run on hardware
//
#include <hip/hip_runtime.h>

#define D_MODEL 256
#define NHEADS  8
#define DH      32
#define KNBR    16
#define BATCH   16
#define NTOK    1024
#define MTOT    (BATCH * NTOK)   // 16384 rows

typedef __attribute__((ext_vector_type(16))) __bf16 v16bf;
typedef __attribute__((ext_vector_type(8)))  float  v8f;
typedef __attribute__((ext_vector_type(8)))  int    v8i;

union Frag { v8i i; v16bf b; };

__device__ __forceinline__ unsigned pack2bf16(float a, float b) {
  unsigned ua = __float_as_uint(a);
  unsigned ub = __float_as_uint(b);
  ua = (ua + 0x7FFFu + ((ua >> 16) & 1u)) >> 16;   // RNE f32 -> bf16
  ub = (ub + 0x7FFFu + ((ub >> 16) & 1u)) >> 16;
  return ua | (ub << 16);
}

// ---------------------------------------------------------------- KNN
__global__ void knn_kernel(const float* __restrict__ pos, int* __restrict__ idx) {
  __shared__ float px[NTOK], py[NTOK], pz[NTOK];
  int b = blockIdx.x >> 2;
  int i = ((blockIdx.x & 3) << 8) + threadIdx.x;
  const float* pb = pos + (size_t)b * NTOK * 3;
  for (int j = threadIdx.x; j < NTOK; j += 256) {
    px[j] = pb[j * 3 + 0];
    py[j] = pb[j * 3 + 1];
    pz[j] = pb[j * 3 + 2];
  }
  __syncthreads();
  float qx = px[i], qy = py[i], qz = pz[i];
  float bd[KNBR]; int bi[KNBR];
#pragma unroll
  for (int p = 0; p < KNBR; ++p) { bd[p] = 3.0e38f; bi[p] = p; }
  for (int j = 0; j < NTOK; ++j) {
    float dx = px[j] - qx, dy = py[j] - qy, dz = pz[j] - qz;
    float d2 = dx * dx + dy * dy + dz * dz;
    if (d2 < bd[KNBR - 1]) {
      float cd = d2; int ci = j;
#pragma unroll
      for (int p = 0; p < KNBR; ++p) {
        if (cd < bd[p]) {
          float td = bd[p]; int ti = bi[p];
          bd[p] = cd; bi[p] = ci; cd = td; ci = ti;
        }
      }
    }
  }
  int* op = idx + ((size_t)b * NTOK + i) * KNBR;
#pragma unroll
  for (int p = 0; p < KNBR; ++p) op[p] = bi[p];
}

// ---------------------------------------------------------- PE + x add
__global__ void posenc_kernel(const float* __restrict__ x, const float* __restrict__ xpos,
                              const unsigned char* __restrict__ mask, float* __restrict__ xq) {
  int t = blockIdx.x * blockDim.x + threadIdx.x;   // [0, MTOT*128)
  int row = t >> 7;
  if (row >= MTOT) return;
  int m = t & 127;
  float posx = xpos[row * 3 + 0], posy = xpos[row * 3 + 1];
  int mm = (m < 64) ? m : (m - 64);
  float pos = (m < 64) ? posy : posx;
  // 2*pi * exp(-ln(10000)/64 * mm)
  float r = 6.283185307179586f * __expf(-0.14391156831212787f * (float)mm);
  float s, c;
  __sincosf(pos * r, &s, &c);
  if (mask[row] == 0) { s = 0.f; c = 0.f; }
  int d0 = (m < 64) ? (m << 1) : (128 + (mm << 1));
  size_t base = (size_t)row * 256;
  xq[base + d0]     = x[base + d0] + s;
  xq[base + d0 + 1] = x[base + d0 + 1] + c;
}

// ------------------------------------------------ pack fp32 -> bf16 A-frag
// dst dword index = ((rowblk*8 + s)*32 + lane)*8 + v
// row = rowblk*16 + lane%16 ; K = s*32 + (v>>2)*16 + (lane>>4)*8 + (v&3)*2
__global__ void pack_act_kernel(const float* __restrict__ src, unsigned* __restrict__ dst) {
  int t = blockIdx.x * blockDim.x + threadIdx.x;   // [0, MTOT*128)
  int v = t & 7;
  int lane = (t >> 3) & 31;
  int s = (t >> 8) & 7;
  int rowblk = t >> 11;
  if (rowblk >= MTOT / 16) return;
  int row = (rowblk << 4) + (lane & 15);
  int kb = (s << 5) + ((v >> 2) << 4) + ((lane >> 4) << 3) + ((v & 3) << 1);
  const float* sp = src + (size_t)row * 256 + kb;
  dst[t] = pack2bf16(sp[0], sp[1]);
}

// ------------------------------------------------ pack fp32 -> bf16 B-frag
// B tile is 32(K) x 16(N): K = (v>>2)*16 + (lane>>4)*8 + (v&3)*2 (+0,+1),
// matching the A-frag K order.
__global__ void pack_w_kernel(const float* __restrict__ w, unsigned* __restrict__ dst) {
  int t = blockIdx.x * blockDim.x + threadIdx.x;   // [0, 32768)
  if (t >= 32768) return;
  int v = t & 7;
  int lane = (t >> 3) & 31;
  int s = (t >> 8) & 7;
  int nt = t >> 11;                                 // 0..15
  int n = (nt << 4) + (lane & 15);
  int kb = (s << 5) + ((v >> 2) << 4) + ((lane >> 4) << 3) + ((v & 3) << 1);
  dst[t] = pack2bf16(w[(size_t)kb * 256 + n], w[(size_t)(kb + 1) * 256 + n]);
}

// ------------------------------------------------------- GEMM strip core
// One wave computes a 16-row strip against one 256x256 weight-fragment
// array. Statically-known global pointers (inlined per call site) keep the
// codegen on global_load_b128 with scalar base addressing (no flat ops).
// Rolled tile loop + per-tile batched B loads: one clause of 16 b128 loads,
// partial s_wait_loadcnt overlaps the 8-WMMA chain; no spills (~150 VGPRs).
__device__ __forceinline__ void gemm_strip(const unsigned* __restrict__ ab,
                                           const unsigned* __restrict__ wb,
                                           float* __restrict__ o,
                                           int rowblk, int lane) {
  Frag a[8];
#pragma unroll
  for (int s = 0; s < 8; ++s)
    a[s].i = *(const v8i*)(ab + (size_t)rowblk * 2048 + s * 256 + lane * 8);
#pragma unroll 1
  for (int nt = 0; nt < 16; ++nt) {
    Frag bfr[8];
#pragma unroll
    for (int s = 0; s < 8; ++s)
      bfr[s].i = *(const v8i*)(wb + (size_t)nt * 2048 + s * 256 + lane * 8);
    v8f acc = {};
#pragma unroll
    for (int s = 0; s < 8; ++s)
      acc = __builtin_amdgcn_wmma_f32_16x16x32_bf16(false, a[s].b, false, bfr[s].b,
                                                    (short)0, acc, false, false);
    int row0 = rowblk * 16 + ((lane >> 4) << 3);    // +8 for upper lane half
    int col = nt * 16 + (lane & 15);
#pragma unroll
    for (int r = 0; r < 8; ++r)
      o[(size_t)(row0 + r) * 256 + col] = acc[r];
  }
}

// ------------------------------------------------------------ q/k/v GEMM
__global__ void qkv_gemm_kernel(const unsigned* __restrict__ aq, const unsigned* __restrict__ av,
                                const unsigned* __restrict__ wqf, const unsigned* __restrict__ wkf,
                                const unsigned* __restrict__ wvf,
                                float* __restrict__ q, float* __restrict__ k, float* __restrict__ vo) {
  int wave = threadIdx.x >> 5;
  int lane = threadIdx.x & 31;
  int rowblk = blockIdx.x * 8 + wave;               // 0..1023
  gemm_strip(aq, wqf, q,  rowblk, lane);
  gemm_strip(aq, wkf, k,  rowblk, lane);
  gemm_strip(av, wvf, vo, rowblk, lane);
}

// ----------------------------------------------------------- attention
__global__ void attn_kernel(const float* __restrict__ q, const float* __restrict__ k,
                            const float* __restrict__ v, const int* __restrict__ idx,
                            const unsigned char* __restrict__ mask, float* __restrict__ out) {
  __shared__ float qs[4][256];
  __shared__ float wsm[4][128];
  __shared__ int   is[4][16];
  __shared__ float inv[4][16];
  int w = threadIdx.x >> 5;
  int lane = threadIdx.x & 31;
  int qrow = blockIdx.x * 4 + w;
  if (lane < 16) {
    int id = idx[(size_t)qrow * 16 + lane];
    is[w][lane] = id;
    int b = qrow >> 10;
    inv[w][lane] = mask[(b << 10) + id] ? 0.0f : 1.0f;
  }
#pragma unroll
  for (int j = 0; j < 8; ++j)
    qs[w][j * 32 + lane] = q[(size_t)qrow * 256 + j * 32 + lane];
  __syncthreads();
#pragma unroll
  for (int p = 0; p < 4; ++p) {
    int h = p * 2 + (lane >> 4);
    int n = lane & 15;
    int id = is[w][n];
    const float* kp = k + (size_t)id * 256 + h * 32;
    const float* qp = &qs[w][h * 32];
    float sc = 0.f;
#pragma unroll
    for (int d = 0; d < 32; ++d) sc += qp[d] * kp[d];
    sc *= 0.17677669529663687f;                      // 1/sqrt(32)
    if (inv[w][n] != 0.0f) sc = -1.0e9f;
    float mx = sc;
#pragma unroll
    for (int off = 8; off >= 1; off >>= 1) mx = fmaxf(mx, __shfl_xor(mx, off, 32));
    float e = __expf(sc - mx);
    float sum = e;
#pragma unroll
    for (int off = 8; off >= 1; off >>= 1) sum += __shfl_xor(sum, off, 32);
    wsm[w][h * 16 + n] = e / sum;
  }
  __syncthreads();
#pragma unroll
  for (int j = 0; j < 8; ++j) {                      // head j, dim = lane
    float o = 0.f;
#pragma unroll
    for (int n = 0; n < KNBR; ++n) {
      int id = is[w][n];
      o += wsm[w][j * 16 + n] * v[(size_t)id * 256 + j * 32 + lane];
    }
    out[(size_t)qrow * 256 + j * 32 + lane] = o;
  }
}

// ----------------------------------------------------- output GEMM (Wo)
// attn @ Wo + residual x, masked-zero for invalid rows, writes d_out.
__global__ void out_gemm_kernel(const unsigned* __restrict__ ao, const unsigned* __restrict__ wof,
                                const float* __restrict__ x, const unsigned char* __restrict__ mask,
                                float* __restrict__ out) {
  int wave = threadIdx.x >> 5;
  int lane = threadIdx.x & 31;
  int rowblk = blockIdx.x * 8 + wave;
  Frag a[8];
#pragma unroll
  for (int s = 0; s < 8; ++s)
    a[s].i = *(const v8i*)(ao + (size_t)rowblk * 2048 + s * 256 + lane * 8);
#pragma unroll 1
  for (int nt = 0; nt < 16; ++nt) {
    Frag bfr[8];
#pragma unroll
    for (int s = 0; s < 8; ++s)
      bfr[s].i = *(const v8i*)(wof + (size_t)nt * 2048 + s * 256 + lane * 8);
    v8f acc = {};
#pragma unroll
    for (int s = 0; s < 8; ++s)
      acc = __builtin_amdgcn_wmma_f32_16x16x32_bf16(false, a[s].b, false, bfr[s].b,
                                                    (short)0, acc, false, false);
    int row0 = rowblk * 16 + ((lane >> 4) << 3);
    int col = nt * 16 + (lane & 15);
#pragma unroll
    for (int r = 0; r < 8; ++r) {
      int row = row0 + r;
      float val = mask[row] ? (acc[r] + x[(size_t)row * 256 + col]) : 0.0f;
      out[(size_t)row * 256 + col] = val;
    }
  }
}

// ---------------------------------------------------------------- launch
extern "C" void kernel_launch(void* const* d_in, const int* in_sizes, int n_in,
                              void* d_out, int out_size, void* d_ws, size_t ws_size,
                              hipStream_t stream) {
  const float* x    = (const float*)d_in[0];
  const float* xpos = (const float*)d_in[1];
  const unsigned char* mask = (const unsigned char*)d_in[2];
  const float* Wq = (const float*)d_in[3];
  const float* Wk = (const float*)d_in[4];
  const float* Wv = (const float*)d_in[5];
  const float* Wo = (const float*)d_in[6];

  char* ws = (char*)d_ws;
  size_t off = 0;
  auto alloc = [&](size_t bytes) -> char* {
    char* p = ws + off;
    off += (bytes + 255) & ~(size_t)255;
    return p;
  };
  float*    XQ  = (float*)   alloc((size_t)MTOT * 256 * 4);  // x+pe; reused as ATT
  unsigned* AQ  = (unsigned*)alloc((size_t)MTOT * 128 * 4);  // A-frags; reused as AO
  unsigned* AV  = (unsigned*)alloc((size_t)MTOT * 128 * 4);
  float*    Q   = (float*)   alloc((size_t)MTOT * 256 * 4);
  float*    Kf  = (float*)   alloc((size_t)MTOT * 256 * 4);
  float*    V   = (float*)   alloc((size_t)MTOT * 256 * 4);
  unsigned* WQF = (unsigned*)alloc(32768 * 4);
  unsigned* WKF = (unsigned*)alloc(32768 * 4);
  unsigned* WVF = (unsigned*)alloc(32768 * 4);
  unsigned* WOF = (unsigned*)alloc(32768 * 4);
  int*      IDX = (int*)     alloc((size_t)MTOT * KNBR * 4);
  float*    ATT = XQ;       // reuse: XQ consumed by pack_act before attn runs
  unsigned* AO  = AQ;       // reuse: AQ consumed by qkv_gemm before repack

  knn_kernel   <<<BATCH * 4, 256, 0, stream>>>(xpos, IDX);
  posenc_kernel<<<MTOT * 128 / 256, 256, 0, stream>>>(x, xpos, mask, XQ);
  pack_act_kernel<<<MTOT * 128 / 256, 256, 0, stream>>>(XQ, AQ);
  pack_act_kernel<<<MTOT * 128 / 256, 256, 0, stream>>>(x, AV);
  pack_w_kernel<<<128, 256, 0, stream>>>(Wq, WQF);
  pack_w_kernel<<<128, 256, 0, stream>>>(Wk, WKF);
  pack_w_kernel<<<128, 256, 0, stream>>>(Wv, WVF);
  pack_w_kernel<<<128, 256, 0, stream>>>(Wo, WOF);
  qkv_gemm_kernel<<<128, 256, 0, stream>>>(AQ, AV, WQF, WKF, WVF, Q, Kf, V);
  attn_kernel<<<MTOT / 4, 128, 0, stream>>>(Q, Kf, V, IDX, mask, ATT);
  pack_act_kernel<<<MTOT * 128 / 256, 256, 0, stream>>>(ATT, AO);
  out_gemm_kernel<<<128, 256, 0, stream>>>(AO, WOF, x, mask, (float*)d_out);
}
